// DualAttention_44736379355582
// MI455X (gfx1250) — matrix operand order repaired and checked by reading, not compiled, not run on hardware
//
#include <hip/hip_runtime.h>
#include <hip/hip_bf16.h>

typedef _Float16 v16h __attribute__((ext_vector_type(16)));
typedef _Float16 v8h  __attribute__((ext_vector_type(8)));
typedef float    v8f  __attribute__((ext_vector_type(8)));

constexpr int BM = 128;
constexpr int BN = 128;
constexpr int BK = 32;
constexpr int LSTR = BK + 8;   // LDS row stride (halves): 40 -> 80B rows, 16B aligned chunks

__device__ inline v8h zero_v8h() {
    v8h z;
#pragma unroll
    for (int i = 0; i < 8; ++i) z[i] = (_Float16)0.f;
    return z;
}
__device__ inline v8f zero_v8f() {
    v8f z;
#pragma unroll
    for (int i = 0; i < 8; ++i) z[i] = 0.f;
    return z;
}

// Async DMA: 32 contiguous bytes global -> LDS per lane (two b128 ops, shared addr pair,
// IOFFSET applies to both global and LDS address per CDNA5 ISA 08_async_tensor §4.4).
__device__ inline void async_copy_32B(const _Float16* __restrict__ gsrc, _Float16* ldst) {
    unsigned long long ga = (unsigned long long)gsrc;
    unsigned           la = (unsigned)(uintptr_t)ldst;   // low 32 bits = LDS byte offset
    asm volatile("global_load_async_to_lds_b128 %0, %1, off"           :: "v"(la), "v"(ga) : "memory");
    asm volatile("global_load_async_to_lds_b128 %0, %1, off offset:16" :: "v"(la), "v"(ga) : "memory");
}

// EPI: 0 = store f16, 1 = store f32, 2 = out = alpha*acc + 2*x (f32), 3 = out += alpha*acc (f32)
template<int TA, int TB, int EPI>
__global__ __launch_bounds__(256)
void gemm_wmma_kernel(const _Float16* __restrict__ A, const _Float16* __restrict__ B,
                      void* __restrict__ Cout, const float* __restrict__ Xres,
                      const float* __restrict__ alpha_ptr,
                      int M, int N, int K, int lda, int ldb, int ldc,
                      long long sA, long long sB, long long sC, long long sX) {
    __shared__ _Float16 As[BM * LSTR];
    __shared__ _Float16 Bs[BN * LSTR];   // stored [n][k] (k-major) for vectorized fragment loads

    const int tid  = threadIdx.x;
    const int lane = tid & 31;
    const int wave = tid >> 5;
    const int waveRow = wave & 3;   // 4 wave-rows * 32 rows
    const int waveCol = wave >> 2;  // 2 wave-cols * 64 cols
    const int nBase = blockIdx.x * BN;
    const int mBase = blockIdx.y * BM;

    A += (size_t)blockIdx.z * (size_t)sA;
    B += (size_t)blockIdx.z * (size_t)sB;
    _Float16* C16 = (_Float16*)Cout + (size_t)blockIdx.z * (size_t)sC;
    float*    C32 = (float*)Cout    + (size_t)blockIdx.z * (size_t)sC;
    const float* Xp = Xres ? (Xres + (size_t)blockIdx.z * (size_t)sX) : nullptr;

    v8f acc[2][4];
#pragma unroll
    for (int tm = 0; tm < 2; ++tm)
#pragma unroll
        for (int tn = 0; tn < 4; ++tn) acc[tm][tn] = zero_v8f();

    for (int k0 = 0; k0 < K; k0 += BK) {
        // ---- stage A tile -> As[m][k] ----
        if (TA == 0) {
            // k-contiguous in both global and LDS: pure async DMA to LDS (ASYNCcnt path)
            int m    = tid >> 1;            // 0..127
            int kOff = (tid & 1) * 16;      // 0 or 16
            const _Float16* src = A + (size_t)(mBase + m) * lda + k0 + kOff;
            _Float16* dst = &As[m * LSTR + kOff];
            if (mBase + m < M) {
                async_copy_32B(src, dst);
            } else {
                *(v8h*)dst       = zero_v8h();
                *(v8h*)(dst + 8) = zero_v8h();
            }
        } else {
            int k    = tid >> 3;            // 0..31
            int mOff = (tid & 7) * 16;      // 0..112
            const _Float16* src = A + (size_t)(k0 + k) * lda + mBase + mOff;
            if (k0 + BK < K) __builtin_prefetch(src + (size_t)BK * lda, 0, 0);
#pragma unroll
            for (int i = 0; i < 16; ++i) {
                _Float16 v = (mBase + mOff + i < M) ? src[i] : (_Float16)0.f;
                As[(mOff + i) * LSTR + k] = v;
            }
        }
        // ---- stage B tile -> Bs[n][k] ----
        if (TB == 0) {
            int k    = tid >> 3;            // 0..31
            int nOff = (tid & 7) * 16;
            const _Float16* src = B + (size_t)(k0 + k) * ldb + nBase + nOff;
            if (k0 + BK < K) __builtin_prefetch(src + (size_t)BK * ldb, 0, 0);
#pragma unroll
            for (int c = 0; c < 2; ++c) {
                int nb = nBase + nOff + c * 8;
                v8h r = zero_v8h();
                if (nb + 8 <= N) {
                    r = *(const v8h*)(src + c * 8);
                } else {
#pragma unroll
                    for (int i = 0; i < 8; ++i)
                        if (nb + i < N) r[i] = src[c * 8 + i];
                }
#pragma unroll
                for (int i = 0; i < 8; ++i)
                    Bs[(nOff + c * 8 + i) * LSTR + k] = r[i];
            }
        } else {
            // k-contiguous in both global and LDS: async DMA to LDS
            int n    = tid >> 1;            // 0..127
            int kOff = (tid & 1) * 16;
            const _Float16* src = B + (size_t)(nBase + n) * ldb + k0 + kOff;
            _Float16* dst = &Bs[n * LSTR + kOff];
            if (nBase + n < N) {
                async_copy_32B(src, dst);
            } else {
                *(v8h*)dst       = zero_v8h();
                *(v8h*)(dst + 8) = zero_v8h();
            }
        }
        // Async LDS writes must complete before the workgroup barrier releases readers.
        if (TA == 0 || TB == 1) asm volatile("s_wait_asynccnt 0x0" ::: "memory");
        __syncthreads();

        // ---- fragments (ISA 16-bit A 16x32 / B 32x16 lane layouts) ----
        const int laneM = lane & 15;
        const int kHiA  = (lane >> 4) * 8;    // A: k = (e/8)*16 + (lane/16)*8 + (e%8)
        const int kHiB  = (lane >> 4) * 8;    // B: k = (e/8)*16 + (lane/16)*8 + (e%8)
        v16h aF[2], bF[4];
#pragma unroll
        for (int tm = 0; tm < 2; ++tm) {
            const _Float16* row = &As[(waveRow * 32 + tm * 16 + laneM) * LSTR];
            v8h lo = *(const v8h*)(row + kHiA);
            v8h hi = *(const v8h*)(row + 16 + kHiA);
            aF[tm] = __builtin_shufflevector(lo, hi, 0,1,2,3,4,5,6,7,8,9,10,11,12,13,14,15);
        }
#pragma unroll
        for (int tn = 0; tn < 4; ++tn) {
            const _Float16* row = &Bs[(waveCol * 64 + tn * 16 + laneM) * LSTR + kHiB];
            v8h lo = *(const v8h*)(row);
            v8h hi = *(const v8h*)(row + 16);
            bF[tn] = __builtin_shufflevector(lo, hi, 0,1,2,3,4,5,6,7,8,9,10,11,12,13,14,15);
        }
#pragma unroll
        for (int tm = 0; tm < 2; ++tm)
#pragma unroll
            for (int tn = 0; tn < 4; ++tn)
                acc[tm][tn] = __builtin_amdgcn_wmma_f32_16x16x32_f16(
                    false, aF[tm], false, bF[tn], (short)0, acc[tm][tn], false, false);
        __syncthreads();
    }

    // ---- epilogue ----
    float alpha = (EPI >= 2) ? alpha_ptr[0] : 0.f;
    const int laneM = lane & 15;
    const int mHi   = (lane >> 4) * 8;   // C/D: m = r + (lane/16)*8, n = lane%16
#pragma unroll
    for (int tm = 0; tm < 2; ++tm) {
#pragma unroll
        for (int tn = 0; tn < 4; ++tn) {
#pragma unroll
            for (int r = 0; r < 8; ++r) {
                int m = mBase + waveRow * 32 + tm * 16 + r + mHi;
                int n = nBase + waveCol * 64 + tn * 16 + laneM;
                if (m < M && n < N) {
                    size_t off = (size_t)m * ldc + n;
                    float v = acc[tm][tn][r];
                    if (EPI == 0)      C16[off] = (_Float16)v;
                    else if (EPI == 1) C32[off] = v;
                    else if (EPI == 2) C32[off] = alpha * v + 2.0f * Xp[off];
                    else               C32[off] += alpha * v;
                }
            }
        }
    }
}

__global__ __launch_bounds__(256)
void softmax_rows_kernel(const float* __restrict__ S, _Float16* __restrict__ P, int cols) {
    __shared__ float red[256];
    size_t row = blockIdx.x;
    const float* s = S + row * (size_t)cols;
    _Float16*    p = P + row * (size_t)cols;
    int t = threadIdx.x;

    float mx = -3.402823466e38f;
    for (int c = t; c < cols; c += 256) mx = fmaxf(mx, s[c]);
    red[t] = mx; __syncthreads();
    for (int o = 128; o > 0; o >>= 1) { if (t < o) red[t] = fmaxf(red[t], red[t + o]); __syncthreads(); }
    mx = red[0]; __syncthreads();

    float sum = 0.f;
    for (int c = t; c < cols; c += 256) sum += __expf(s[c] - mx);
    red[t] = sum; __syncthreads();
    for (int o = 128; o > 0; o >>= 1) { if (t < o) red[t] += red[t + o]; __syncthreads(); }
    float inv = 1.0f / red[0];

    for (int c = t; c < cols; c += 256) p[c] = (_Float16)(__expf(s[c] - mx) * inv);
}

__global__ __launch_bounds__(256)
void f32_to_f16_kernel(const float* __restrict__ in, _Float16* __restrict__ out, size_t n) {
    size_t i = (size_t)blockIdx.x * 256 + threadIdx.x;
    size_t stride = (size_t)gridDim.x * 256;
    for (; i < n; i += stride) out[i] = (_Float16)in[i];
}

extern "C" void kernel_launch(void* const* d_in, const int* in_sizes, int n_in,
                              void* d_out, int out_size, void* d_ws, size_t ws_size,
                              hipStream_t stream) {
    (void)in_sizes; (void)n_in; (void)out_size; (void)ws_size;
    const int  Bb = 16, C = 1280, d = 160, N = 1024;

    const float* x     = (const float*)d_in[0];
    const float* Wk    = (const float*)d_in[1];
    const float* Wq    = (const float*)d_in[2];
    const float* Wv    = (const float*)d_in[3];
    const float* gamma = (const float*)d_in[4];
    const float* beta  = (const float*)d_in[5];
    float* out = (float*)d_out;

    // workspace layout (256B-aligned)
    char* base = (char*)d_ws;
    size_t off = 0;
    auto alloc = [&](size_t bytes) { char* p = base + off; off = (off + bytes + 255) & ~(size_t)255; return p; };
    _Float16* xh  = (_Float16*)alloc((size_t)Bb * N * C * 2);
    _Float16* vh  = (_Float16*)alloc((size_t)Bb * N * C * 2);
    _Float16* kh  = (_Float16*)alloc((size_t)Bb * N * d * 2);
    _Float16* qh  = (_Float16*)alloc((size_t)Bb * N * d * 2);
    _Float16* wkh = (_Float16*)alloc((size_t)C * d * 2);
    _Float16* wqh = (_Float16*)alloc((size_t)C * d * 2);
    _Float16* wvh = (_Float16*)alloc((size_t)C * C * 2);
    float*    sbuf = (float*)   alloc((size_t)Bb * C * C * 4);   // reused for s [N*N] and s2 [C*C]
    _Float16* pbuf = (_Float16*)alloc((size_t)Bb * C * C * 2);   // reused for p and p2

    // ---- f32 -> f16 conversions ----
    f32_to_f16_kernel<<<4096, 256, 0, stream>>>(x,  xh,  (size_t)Bb * N * C);
    f32_to_f16_kernel<<<256,  256, 0, stream>>>(Wk, wkh, (size_t)C * d);
    f32_to_f16_kernel<<<256,  256, 0, stream>>>(Wq, wqh, (size_t)C * d);
    f32_to_f16_kernel<<<1024, 256, 0, stream>>>(Wv, wvh, (size_t)C * C);

    dim3 blk(256);
    // ---- k = xh @ Wk  [N x d] ----
    gemm_wmma_kernel<0,0,0><<<dim3((d + 127) / 128, N / 128, Bb), blk, 0, stream>>>(
        xh, wkh, kh, nullptr, nullptr, N, d, C, C, d, d,
        (long long)N * C, 0, (long long)N * d, 0);
    // ---- q = xh @ Wq ----
    gemm_wmma_kernel<0,0,0><<<dim3((d + 127) / 128, N / 128, Bb), blk, 0, stream>>>(
        xh, wqh, qh, nullptr, nullptr, N, d, C, C, d, d,
        (long long)N * C, 0, (long long)N * d, 0);
    // ---- v = xh @ Wv  [N x C] ----
    gemm_wmma_kernel<0,0,0><<<dim3(C / 128, N / 128, Bb), blk, 0, stream>>>(
        xh, wvh, vh, nullptr, nullptr, N, C, C, C, C, C,
        (long long)N * C, 0, (long long)N * C, 0);
    // ---- s = k @ q^T  [N x N] f32 ----
    gemm_wmma_kernel<0,1,1><<<dim3(N / 128, N / 128, Bb), blk, 0, stream>>>(
        kh, qh, sbuf, nullptr, nullptr, N, N, d, d, d, N,
        (long long)N * d, (long long)N * d, (long long)N * N, 0);
    // ---- p = softmax(s) -> f16 ----
    softmax_rows_kernel<<<Bb * N, 256, 0, stream>>>(sbuf, pbuf, N);
    // ---- out = gamma * (p @ v) + 2*x ----
    gemm_wmma_kernel<0,0,2><<<dim3(C / 128, N / 128, Bb), blk, 0, stream>>>(
        pbuf, vh, out, x, gamma, N, C, N, N, C, C,
        (long long)N * N, (long long)N * C, (long long)N * C, (long long)N * C);
    // ---- s2 = xh^T @ xh  [C x C] f32 ----
    gemm_wmma_kernel<1,0,1><<<dim3(C / 128, C / 128, Bb), blk, 0, stream>>>(
        xh, xh, sbuf, nullptr, nullptr, C, C, N, C, C, C,
        (long long)N * C, (long long)N * C, (long long)C * C, 0);
    // ---- p2 = softmax(s2) -> f16 ----
    softmax_rows_kernel<<<Bb * C, 256, 0, stream>>>(sbuf, pbuf, C);
    // ---- out += beta * (xh @ p2) ----
    gemm_wmma_kernel<0,0,3><<<dim3(C / 128, N / 128, Bb), blk, 0, stream>>>(
        xh, pbuf, out, nullptr, beta, N, C, C, C, C, C,
        (long long)N * C, (long long)C * C, (long long)N * C, 0);
}
